// MemoryUpdaterModule_38474317037915
// MI455X (gfx1250) — matrix operand order repaired and checked by reading, not compiled, not run on hardware
//
#include <hip/hip_runtime.h>
#include <hip/hip_bf16.h>
#include <math.h>

#define Bb   32
#define Ss   512
#define Dd   1024
#define Hh   16
#define DHh  64
#define Ll   4
#define DFFf 2048
#define DMEMm 512
#define SBr  (Ss*Bb)      /* 16384 rows, layout [s*B+b, D] */
#define D3   (3*Dd)

typedef __attribute__((ext_vector_type(16))) __bf16 bf16x16;
typedef __attribute__((ext_vector_type(8)))  __bf16 bf16x8;
typedef __attribute__((ext_vector_type(8)))  float  f32x8;

union ABfrag { bf16x16 v; bf16x8 h[2]; __bf16 e[16]; };

__device__ __forceinline__ f32x8 wmma_bf16(const ABfrag& a, const ABfrag& b, f32x8 c) {
    // D = A(16x32 bf16) * B(32x16 bf16) + C(16x16 f32)
    return __builtin_amdgcn_wmma_f32_16x16x32_bf16(false, a.v, false, b.v, (short)0, c,
                                                   false, false);
}

__device__ __forceinline__ f32x8 zero8() {
    f32x8 z = {0.f, 0.f, 0.f, 0.f, 0.f, 0.f, 0.f, 0.f};
    return z;
}

// ---------------------------------------------------------------------------
// fp32 -> bf16 conversion
// ---------------------------------------------------------------------------
__global__ __launch_bounds__(256)
void tr_f32_to_bf16(const float* __restrict__ s, __bf16* __restrict__ d, int n) {
    int i = blockIdx.x * 256 + threadIdx.x;
    if (i < n) d[i] = (__bf16)s[i];
}

// ---------------------------------------------------------------------------
// h[s*B+b, :] = x[b, s, :] + PE(s, :)   (f32 + bf16 copies)
// ---------------------------------------------------------------------------
__global__ __launch_bounds__(256)
void tr_addpe(const float* __restrict__ x, float* __restrict__ h, __bf16* __restrict__ hbf) {
    int i = blockIdx.x * 256 + threadIdx.x;
    if (i >= SBr * (Dd / 2)) return;
    int dh  = i % (Dd / 2);
    int row = i / (Dd / 2);          // row = s*B + b
    int bbb = row & (Bb - 1);
    int sss = row >> 5;              // B == 32
    const size_t xi = ((size_t)bbb * Ss + sss) * Dd + 2 * dh;
    float div = __expf((float)(2 * dh) * (-9.210340371976184f / (float)Dd)); // ln(10000)
    float ang = (float)sss * div;
    float ev = x[xi]     + __sinf(ang);
    float ov = x[xi + 1] + __cosf(ang);
    const size_t hi = (size_t)row * Dd + 2 * dh;
    h[hi]     = ev;  h[hi + 1]   = ov;
    hbf[hi]   = (__bf16)ev;
    hbf[hi+1] = (__bf16)ov;
}

// ---------------------------------------------------------------------------
// C = act(A[M,K]bf16 @ W[N,K]bf16^T + bias)
// 8 waves/WG; wave => 32 rows x 64 cols (2x4 wmma 16x16 accumulators),
// software-pipelined K loop (double-buffered fragments).
// flags: 1=relu, 2=write f32, 4=write bf16, 8=remap rows s*B+b -> b*S+s
// ---------------------------------------------------------------------------
struct Frags { ABfrag a[2]; ABfrag b[4]; };

__device__ __forceinline__ void tr_load_frags(Frags& f,
                                              const __bf16* __restrict__ A,
                                              const __bf16* __restrict__ W,
                                              size_t ar0, size_t ar1,
                                              const size_t* wr,
                                              int k0, int aoff, int boff) {
    const __bf16* p0 = A + ar0 + k0 + aoff;
    f.a[0].h[0] = *(const bf16x8*)(p0);
    f.a[0].h[1] = *(const bf16x8*)(p0 + 16);
    const __bf16* p1 = A + ar1 + k0 + aoff;
    f.a[1].h[0] = *(const bf16x8*)(p1);
    f.a[1].h[1] = *(const bf16x8*)(p1 + 16);
#pragma unroll
    for (int j = 0; j < 4; j++) {
        const __bf16* wp = W + wr[j] + k0 + boff;
        f.b[j].h[0] = *(const bf16x8*)(wp);
        f.b[j].h[1] = *(const bf16x8*)(wp + 16);
    }
}

__device__ __forceinline__ void tr_mma_frags(f32x8 (&acc)[2][4], const Frags& f) {
#pragma unroll
    for (int i = 0; i < 2; i++)
#pragma unroll
        for (int j = 0; j < 4; j++)
            acc[i][j] = wmma_bf16(f.a[i], f.b[j], acc[i][j]);
}

__global__ __launch_bounds__(256)
void tr_gemm(const __bf16* __restrict__ A, const __bf16* __restrict__ W,
             const float* __restrict__ bias,
             float* __restrict__ Cf, __bf16* __restrict__ Cb,
             int M, int N, int K, int flags) {
    const int lane = threadIdx.x & 31;
    const int wv   = threadIdx.x >> 5;
    const int m0   = blockIdx.x * 256 + wv * 32;
    const int n0   = blockIdx.y * 64;
    const int col  = lane & 15;
    const int half = lane >> 4;
    const int aoff = half ? 8 : 0;    // A frag: lane<16 K{0..7,16..23}, lane>=16 {8..15,24..31}
    const int boff = half ? 8 : 0;    // B frag: lane holds weight row n, K same split as A

    f32x8 acc[2][4];
#pragma unroll
    for (int i = 0; i < 2; i++)
#pragma unroll
        for (int j = 0; j < 4; j++) acc[i][j] = zero8();

    const size_t ar0 = (size_t)(m0 + col) * K;
    const size_t ar1 = (size_t)(m0 + 16 + col) * K;
    size_t wr[4];
#pragma unroll
    for (int j = 0; j < 4; j++) wr[j] = (size_t)(n0 + j * 16 + col) * K;

    // software pipeline: K is always a multiple of 64 here
    Frags f0, f1;
    tr_load_frags(f0, A, W, ar0, ar1, wr, 0, aoff, boff);
    for (int k0 = 0; k0 < K; k0 += 64) {
        tr_load_frags(f1, A, W, ar0, ar1, wr, k0 + 32, aoff, boff);
        tr_mma_frags(acc, f0);
        if (k0 + 64 < K)
            tr_load_frags(f0, A, W, ar0, ar1, wr, k0 + 64, aoff, boff);
        tr_mma_frags(acc, f1);
    }

#pragma unroll
    for (int i = 0; i < 2; i++) {
#pragma unroll
        for (int j = 0; j < 4; j++) {
            const int n  = n0 + j * 16 + col;
            const float bv = bias[n];
#pragma unroll
            for (int r = 0; r < 8; r++) {
                int m   = m0 + i * 16 + r + half * 8;
                float v = acc[i][j][r] + bv;
                if (flags & 1) v = fmaxf(v, 0.f);
                int orow = m;
                if (flags & 8) orow = (m & (Bb - 1)) * Ss + (m >> 5);  // (b,s) remap
                if (flags & 2) Cf[(size_t)orow * N + n] = v;
                if (flags & 4) Cb[(size_t)orow * N + n] = (__bf16)v;
            }
        }
    }
}

// ---------------------------------------------------------------------------
// Flash attention: one wave = one (b, h, 16-row q block)
// qkv layout: [s*B+b, 3D] bf16, Q at col 0, K at col D, V at col 2D
// ---------------------------------------------------------------------------
__global__ __launch_bounds__(128)
void tr_attn(const __bf16* __restrict__ qkv, __bf16* __restrict__ ctx) {
    __shared__ __align__(16) __bf16 pbuf[4 * 16 * 32];

    const int lane = threadIdx.x & 31;
    const int wvb  = threadIdx.x >> 5;
    const int wid  = blockIdx.x * 4 + wvb;
    const int qb   = wid & 31;          // 32 q-blocks of 16
    const int hh   = (wid >> 5) & 15;   // head
    const int bb   = wid >> 9;          // batch
    const int col  = lane & 15;
    const int half = lane >> 4;
    const int q0   = qb * 16;

    __bf16* pl = pbuf + wvb * (16 * 32);

    // Q as A-fragments, K-dim = DH = 64 -> two 16x32 fragments
    ABfrag aq[2];
    {
        const __bf16* Qp = qkv + ((size_t)(q0 + col) * Bb + bb) * D3 + hh * DHh;
#pragma unroll
        for (int kk = 0; kk < 2; kk++) {
            const __bf16* p = Qp + kk * 32 + (half ? 16 : 0);
            aq[kk].h[0] = *(const bf16x8*)(p);
            aq[kk].h[1] = *(const bf16x8*)(p + 8);
        }
    }

    f32x8 o[4];
#pragma unroll
    for (int j = 0; j < 4; j++) o[j] = zero8();
    float mrow[8], lrow[8];
#pragma unroll
    for (int r = 0; r < 8; r++) { mrow[r] = -1e30f; lrow[r] = 0.f; }
    const float scale = 0.125f;  // 1/sqrt(64)

    for (int t0 = 0; t0 < Ss; t0 += 32) {
        // --- scores S = Q @ K^T for 32 key columns (two 16x16 tiles) ---
        f32x8 s0 = zero8(), s1 = zero8();
#pragma unroll
        for (int kk = 0; kk < 2; kk++) {
            const size_t kbase = (size_t)Dd + hh * DHh + kk * 32 + (half ? 16 : 0);
            const __bf16* Kp0 = qkv + ((size_t)(t0 + col) * Bb + bb) * D3 + kbase;
            ABfrag b0; b0.h[0] = *(const bf16x8*)(Kp0); b0.h[1] = *(const bf16x8*)(Kp0 + 8);
            s0 = wmma_bf16(aq[kk], b0, s0);
            const __bf16* Kp1 = qkv + ((size_t)(t0 + 16 + col) * Bb + bb) * D3 + kbase;
            ABfrag b1; b1.h[0] = *(const bf16x8*)(Kp1); b1.h[1] = *(const bf16x8*)(Kp1 + 8);
            s1 = wmma_bf16(aq[kk], b1, s1);
        }
        // --- online softmax (rows live within one 16-lane half) ---
#pragma unroll
        for (int r = 0; r < 8; r++) {
            float v0 = s0[r] * scale, v1 = s1[r] * scale;
            float rm = fmaxf(v0, v1);
            rm = fmaxf(rm, __shfl_xor(rm, 1, 32));
            rm = fmaxf(rm, __shfl_xor(rm, 2, 32));
            rm = fmaxf(rm, __shfl_xor(rm, 4, 32));
            rm = fmaxf(rm, __shfl_xor(rm, 8, 32));
            float mnew  = fmaxf(mrow[r], rm);
            float alpha = __expf(mrow[r] - mnew);
            mrow[r] = mnew;
            float p0 = __expf(v0 - mnew), p1 = __expf(v1 - mnew);
            float rs = p0 + p1;
            rs += __shfl_xor(rs, 1, 32);
            rs += __shfl_xor(rs, 2, 32);
            rs += __shfl_xor(rs, 4, 32);
            rs += __shfl_xor(rs, 8, 32);
            lrow[r] = lrow[r] * alpha + rs;
#pragma unroll
            for (int j = 0; j < 4; j++) o[j][r] *= alpha;
            // stage P (C-tile layout) to LDS row-major for A-fragment reload
            pl[(r + half * 8) * 32 + col]      = (__bf16)p0;
            pl[(r + half * 8) * 32 + 16 + col] = (__bf16)p1;
        }
        asm volatile("" ::: "memory");   // keep DS store->load order (HW LDS is in-order per wave)
        ABfrag pa;
        const __bf16* pp = pl + col * 32 + (half ? 16 : 0);
        pa.h[0] = *(const bf16x8*)(pp);
        pa.h[1] = *(const bf16x8*)(pp + 8);
        asm volatile("" ::: "memory");
        // --- O += P @ V (K = 32 key rows) ---
#pragma unroll
        for (int j = 0; j < 4; j++) {
            const __bf16* Vp = qkv + ((size_t)(t0 + half * 16) * Bb + bb) * D3
                               + 2 * Dd + hh * DHh + j * 16 + col;
            ABfrag bv;
#pragma unroll
            for (int i = 0; i < 16; i++) bv.e[i] = Vp[(size_t)i * Bb * D3];
            o[j] = wmma_bf16(pa, bv, o[j]);
        }
    }
    // --- finalize: O / l, store bf16 context ---
#pragma unroll
    for (int r = 0; r < 8; r++) {
        float inv = 1.f / lrow[r];
        const size_t orow = ((size_t)(q0 + r + half * 8) * Bb + bb) * Dd + hh * DHh;
#pragma unroll
        for (int j = 0; j < 4; j++)
            ctx[orow + j * 16 + col] = (__bf16)(o[j][r] * inv);
    }
}

// ---------------------------------------------------------------------------
// h = LayerNorm(hin (+ resid)) * g + b   -> f32 and bf16
// one 256-thread block per row of 1024
// ---------------------------------------------------------------------------
__device__ __forceinline__ float tr_block_sum(float v) {
#pragma unroll
    for (int m = 16; m >= 1; m >>= 1) v += __shfl_xor(v, m, 32);
    __shared__ float sb[8];
    int w = threadIdx.x >> 5, ln = threadIdx.x & 31;
    if (ln == 0) sb[w] = v;
    __syncthreads();
    if (w == 0) {
        float t = (ln < 8) ? sb[ln] : 0.f;
#pragma unroll
        for (int m = 4; m >= 1; m >>= 1) t += __shfl_xor(t, m, 32);
        if (ln == 0) sb[0] = t;
    }
    __syncthreads();
    v = sb[0];
    __syncthreads();
    return v;
}

__global__ __launch_bounds__(256)
void tr_resid_ln(const float* __restrict__ hin, const float* __restrict__ resid, int addf,
                 const float* __restrict__ g, const float* __restrict__ beta,
                 float* __restrict__ hout, __bf16* __restrict__ hbf) {
    const size_t base = (size_t)blockIdx.x * Dd;
    const int i0 = threadIdx.x * 4;
    float4 x = *(const float4*)(hin + base + i0);
    if (addf) {
        const float4 rr = *(const float4*)(resid + base + i0);
        x.x += rr.x; x.y += rr.y; x.z += rr.z; x.w += rr.w;
    }
    float mean = tr_block_sum(x.x + x.y + x.z + x.w) * (1.f / (float)Dd);
    float dx = x.x - mean, dy = x.y - mean, dz = x.z - mean, dw = x.w - mean;
    float var  = tr_block_sum(dx * dx + dy * dy + dz * dz + dw * dw) * (1.f / (float)Dd);
    float inv  = rsqrtf(var + 1e-5f);
    float4 gv = *(const float4*)(g + i0);
    float4 bv = *(const float4*)(beta + i0);
    float4 y;
    y.x = dx * inv * gv.x + bv.x;
    y.y = dy * inv * gv.y + bv.y;
    y.z = dz * inv * gv.z + bv.z;
    y.w = dw * inv * gv.w + bv.w;
    *(float4*)(hout + base + i0) = y;
    hbf[base + i0 + 0] = (__bf16)y.x;
    hbf[base + i0 + 1] = (__bf16)y.y;
    hbf[base + i0 + 2] = (__bf16)y.z;
    hbf[base + i0 + 3] = (__bf16)y.w;
}

// ---------------------------------------------------------------------------
// Host orchestration
// ---------------------------------------------------------------------------
extern "C" void kernel_launch(void* const* d_in, const int* in_sizes, int n_in,
                              void* d_out, int out_size, void* d_ws, size_t ws_size,
                              hipStream_t stream) {
    (void)in_sizes; (void)n_in; (void)out_size; (void)ws_size;
    const float* x    = (const float*)d_in[0];
    const float* Wqkv = (const float*)d_in[1];
    const float* bqkv = (const float*)d_in[2];
    const float* Wo   = (const float*)d_in[3];
    const float* bo   = (const float*)d_in[4];
    const float* W1   = (const float*)d_in[5];
    const float* b1   = (const float*)d_in[6];
    const float* W2   = (const float*)d_in[7];
    const float* b2   = (const float*)d_in[8];
    const float* ln1g = (const float*)d_in[9];
    const float* ln1b = (const float*)d_in[10];
    const float* ln2g = (const float*)d_in[11];
    const float* ln2b = (const float*)d_in[12];
    const float* nfg  = (const float*)d_in[13];
    const float* nfb  = (const float*)d_in[14];
    const float* Wout = (const float*)d_in[15];
    const float* bout = (const float*)d_in[16];
    float* out = (float*)d_out;

    char* p = (char*)d_ws;
    auto take = [&](size_t bytes) {
        void* r = (void*)p;
        p += (bytes + 255) & ~(size_t)255;
        return r;
    };
    float*  h_f32   = (float*) take((size_t)SBr * Dd * 4);
    __bf16* h_bf    = (__bf16*)take((size_t)SBr * Dd * 2);
    float*  tmp_f32 = (float*) take((size_t)SBr * Dd * 4);
    __bf16* qkv_bf  = (__bf16*)take((size_t)SBr * 3 * Dd * 2);
    __bf16* ctx_bf  = (__bf16*)take((size_t)SBr * Dd * 2);
    __bf16* ff_bf   = (__bf16*)take((size_t)SBr * DFFf * 2);
    __bf16* Wqkv_bf = (__bf16*)take((size_t)Ll * 3 * Dd * Dd * 2);
    __bf16* Wo_bf   = (__bf16*)take((size_t)Ll * Dd * Dd * 2);
    __bf16* W1_bf   = (__bf16*)take((size_t)Ll * DFFf * Dd * 2);
    __bf16* W2_bf   = (__bf16*)take((size_t)Ll * Dd * DFFf * 2);
    __bf16* Wout_bf = (__bf16*)take((size_t)DMEMm * Dd * 2);

    auto conv = [&](const float* s, __bf16* d, size_t n) {
        tr_f32_to_bf16<<<dim3((unsigned)((n + 255) / 256)), dim3(256), 0, stream>>>(s, d, (int)n);
    };
    conv(Wqkv, Wqkv_bf, (size_t)Ll * 3 * Dd * Dd);
    conv(Wo,   Wo_bf,   (size_t)Ll * Dd * Dd);
    conv(W1,   W1_bf,   (size_t)Ll * DFFf * Dd);
    conv(W2,   W2_bf,   (size_t)Ll * Dd * DFFf);
    conv(Wout, Wout_bf, (size_t)DMEMm * Dd);

    tr_addpe<<<dim3(SBr * (Dd / 2) / 256), dim3(256), 0, stream>>>(x, h_f32, h_bf);

    for (int l = 0; l < Ll; l++) {
        // QKV projection -> bf16
        tr_gemm<<<dim3(SBr / 256, (3 * Dd) / 64), dim3(256), 0, stream>>>(
            h_bf, Wqkv_bf + (size_t)l * 3 * Dd * Dd, bqkv + (size_t)l * 3 * Dd,
            nullptr, qkv_bf, SBr, 3 * Dd, Dd, /*bf16*/4);
        // fused flash attention -> ctx bf16
        tr_attn<<<dim3((Bb * Hh * (Ss / 16)) / 4), dim3(128), 0, stream>>>(qkv_bf, ctx_bf);
        // output projection -> f32 residual
        tr_gemm<<<dim3(SBr / 256, Dd / 64), dim3(256), 0, stream>>>(
            ctx_bf, Wo_bf + (size_t)l * Dd * Dd, bo + (size_t)l * Dd,
            tmp_f32, nullptr, SBr, Dd, Dd, /*f32*/2);
        tr_resid_ln<<<dim3(SBr), dim3(256), 0, stream>>>(
            h_f32, tmp_f32, 1, ln1g + (size_t)l * Dd, ln1b + (size_t)l * Dd, h_f32, h_bf);
        // FFN
        tr_gemm<<<dim3(SBr / 256, DFFf / 64), dim3(256), 0, stream>>>(
            h_bf, W1_bf + (size_t)l * DFFf * Dd, b1 + (size_t)l * DFFf,
            nullptr, ff_bf, SBr, DFFf, Dd, /*relu|bf16*/1 | 4);
        tr_gemm<<<dim3(SBr / 256, Dd / 64), dim3(256), 0, stream>>>(
            ff_bf, W2_bf + (size_t)l * Dd * DFFf, b2 + (size_t)l * Dd,
            tmp_f32, nullptr, SBr, Dd, DFFf, /*f32*/2);
        tr_resid_ln<<<dim3(SBr), dim3(256), 0, stream>>>(
            h_f32, tmp_f32, 1, ln2g + (size_t)l * Dd, ln2b + (size_t)l * Dd, h_f32, h_bf);
    }
    // final norm
    tr_resid_ln<<<dim3(SBr), dim3(256), 0, stream>>>(h_f32, h_f32, 0, nfg, nfb, h_f32, h_bf);
    // output projection with [s*B+b] -> [b*S+s] remap, f32 out
    tr_gemm<<<dim3(SBr / 256, DMEMm / 64), dim3(256), 0, stream>>>(
        h_bf, Wout_bf, bout, out, nullptr, SBr, DMEMm, Dd, /*f32|remap*/2 | 8);
}
